// outDCNconv_45586782880227
// MI455X (gfx1250) — compile-verified, matrix operand order checked by reading, not run on hardware
//
#include <hip/hip_runtime.h>
#include <math.h>

typedef __bf16 bf16_t;
typedef __attribute__((ext_vector_type(16))) __bf16        v16bf;
typedef __attribute__((ext_vector_type(8)))  float         v8f;
typedef __attribute__((ext_vector_type(8)))  unsigned int  v8u;

#define HWSZ  65536     // 256*256
#define CCH   64
#define KK    576       // 9 * 64, K ordering: kpos*64 + c
#define NCHK  18        // 576 / 32
#define LDSK  40        // padded LDS row stride (bf16 elems) to avoid bank conflicts
#define NPOS  128       // spatial positions per block tile

// ---------------- bf16 helpers (RNE) ----------------
__device__ __forceinline__ bf16_t f2bf(float f) {
  unsigned u = __float_as_uint(f);
  u += 0x7FFFu + ((u >> 16) & 1u);
  unsigned short h = (unsigned short)(u >> 16);
  return __builtin_bit_cast(bf16_t, h);
}
__device__ __forceinline__ float bf2f(bf16_t b) {
  unsigned short h = __builtin_bit_cast(unsigned short, b);
  return __uint_as_float(((unsigned)h) << 16);
}
__device__ __forceinline__ unsigned pack_bf2(float a, float b) {
  unsigned ua = __float_as_uint(a); ua += 0x7FFFu + ((ua >> 16) & 1u);
  unsigned ub = __float_as_uint(b); ub += 0x7FFFu + ((ub >> 16) & 1u);
  return (ua >> 16) | (ub & 0xFFFF0000u);
}

// ---------------- WMMA fragment helpers ----------------
// 16-bit A/B fragment layout (ISA 05_wmma 7.12.2):
// lane 0-15 : K pairs {0,1}..{6,7} in v0..v3, {16,17}..{22,23} in v4..v7
// lane 16-31: K pairs {8,9}..{14,15} in v0..v3, {24,25}..{30,31} in v4..v7
// caller pre-offsets row by (lane>>4)*8.
__device__ __forceinline__ v16bf load_frag_row(const bf16_t* row) {
  v8u p;
#pragma unroll
  for (int r = 0; r < 8; ++r) {
    int kk = ((r & 4) << 2) + ((r & 3) << 1);
    p[r] = *(const unsigned*)(row + kk);
  }
  return __builtin_bit_cast(v16bf, p);
}

__device__ __forceinline__ v8f wmma_bf16(v16bf a, v16bf b, v8f c) {
  return __builtin_amdgcn_wmma_f32_16x16x32_bf16(false, a, false, b, (short)0, c,
                                                 false, false);
}

// ---------------- register-staged im2col gather (one position/thread) --------
// Each thread owns position n = tid of the 128-wide tile and fetches its 32
// K-values (one tap, 32 channels) for `chunk`, packed as 16 dwords.
// Halo handling: always load from a clamped (valid) address, then zero the
// VALUE with one v_and per dword -- no divergent exec-mask branches.
__device__ __forceinline__ void gather_im2col(unsigned vals[16], const bf16_t* src,
                                              int n0, int chunk, int dil, int tid) {
  const int kpos = chunk >> 1;
  const int c0   = (chunk & 1) << 5;
  const int dy   = (kpos / 3 - 1) * dil;
  const int dx   = (kpos % 3 - 1) * dil;
  int pos = n0 + tid;
  int y = (pos >> 8) + dy;
  int x = (pos & 255) + dx;
  bool inb = ((unsigned)y < 256u) && ((unsigned)x < 256u);
  unsigned msk = inb ? 0xFFFFFFFFu : 0u;
  int yc = min(max(y, 0), 255), xc = min(max(x, 0), 255);
  int spi = (yc << 8) + xc;                   // always in-bounds
  const unsigned short* s16 = (const unsigned short*)src;
#pragma unroll
  for (int kk = 0; kk < 16; ++kk) {
    unsigned lo = (unsigned)s16[(size_t)(c0 + 2 * kk) * HWSZ + spi];
    unsigned hi = (unsigned)s16[(size_t)(c0 + 2 * kk + 1) * HWSZ + spi];
    vals[kk] = (lo | (hi << 16)) & msk;
  }
}

__device__ __forceinline__ void store_btile_row(bf16_t* ldsB, const unsigned vals[16],
                                                int tid) {
  char* row = (char*)(ldsB + tid * LDSK);     // tid*80 bytes, 16B aligned
#pragma unroll
  for (int q = 0; q < 4; ++q) {
    uint4 v;
    v.x = vals[4 * q + 0]; v.y = vals[4 * q + 1];
    v.z = vals[4 * q + 2]; v.w = vals[4 * q + 3];
    *(uint4*)(row + 16 * q) = v;
  }
}

// ---------------- prep: fp32 -> bf16 activation copy ----------------
__global__ void __launch_bounds__(256) cvt_kernel(const float* __restrict__ x,
                                                  bf16_t* __restrict__ xb) {
  int i = (blockIdx.x * 256 + threadIdx.x) << 2;
  float4 f = *(const float4*)(x + i);
  uint2 p; p.x = pack_bf2(f.x, f.y); p.y = pack_bf2(f.z, f.w);
  *(uint2*)((char*)xb + (size_t)i * 2) = p;
}

// ---------------- prep: pack weights [O][kpos*64+c] bf16 + fold BN -----------
__global__ void __launch_bounds__(256) pack_kernel(
    const float* __restrict__ w_om, const float* __restrict__ w_dcn,
    const float* __restrict__ w_h,  const float* __restrict__ w_w,
    const float* __restrict__ g1, const float* __restrict__ be1,
    const float* __restrict__ mu1, const float* __restrict__ v1,
    const float* __restrict__ g2, const float* __restrict__ be2,
    const float* __restrict__ mu2, const float* __restrict__ v2,
    const float* __restrict__ g3, const float* __restrict__ be3,
    const float* __restrict__ mu3, const float* __restrict__ v3,
    bf16_t* __restrict__ wbom, bf16_t* __restrict__ wbdcn,
    bf16_t* __restrict__ wbh,  bf16_t* __restrict__ wbw,
    float* __restrict__ bn) {
  int i = blockIdx.x * 256 + threadIdx.x;
  if (i < 32 * KK) {
    int o = i / KK, K = i % KK, kpos = K >> 6, c = K & 63;
    wbom[i] = (o < 27) ? f2bf(w_om[((size_t)o * 64 + c) * 9 + kpos])
                       : __builtin_bit_cast(bf16_t, (unsigned short)0);
    return;
  }
  i -= 32 * KK;
  if (i < 3 * 64 * KK) {
    int which = i / (64 * KK);
    int j = i % (64 * KK);
    int o = j / KK, K = j % KK, kpos = K >> 6, c = K & 63;
    const float* src = (which == 0) ? w_dcn : (which == 1) ? w_h : w_w;
    bf16_t* dst      = (which == 0) ? wbdcn : (which == 1) ? wbh : wbw;
    dst[j] = f2bf(src[((size_t)o * 64 + c) * 9 + kpos]);
    return;
  }
  i -= 3 * 64 * KK;
  if (i < 192) {
    int set = i >> 6, c = i & 63;
    const float* g  = (set == 0) ? g1  : (set == 1) ? g2  : g3;
    const float* be = (set == 0) ? be1 : (set == 1) ? be2 : be3;
    const float* mu = (set == 0) ? mu1 : (set == 1) ? mu2 : mu3;
    const float* vv = (set == 0) ? v1  : (set == 1) ? v2  : v3;
    float s = g[c] * rsqrtf(vv[c] + 1e-5f);
    bn[set * 128 + c]      = s;
    bn[set * 128 + 64 + c] = be[c] - mu[c] * s;
  }
}

// ---------------- offset/mask conv: M=32(27), N-tile 128, double-buffered ----
__global__ void __launch_bounds__(128) gemm_om_kernel(const bf16_t* __restrict__ xb,
                                                      const bf16_t* __restrict__ wb,
                                                      const float* __restrict__ bias,
                                                      float* __restrict__ om) {
  __shared__ bf16_t ldsB[2][NPOS * LDSK];
  const int tid = threadIdx.x, lane = tid & 31, wave = tid >> 5;
  const int m0    = (wave & 1) << 4;       // 2 M-strips
  const int nbase = (wave >> 1) << 6;      // wave pairs split the 128 columns
  const int b  = blockIdx.y;
  const int n0 = blockIdx.x << 7;
  const bf16_t* src = xb + (size_t)b * CCH * HWSZ;
  const bf16_t* arowbase = wb + (size_t)(m0 + (lane & 15)) * KK + ((lane >> 4) << 3);

  v8f zero = {0.f, 0.f, 0.f, 0.f, 0.f, 0.f, 0.f, 0.f};
  v8f acc[4] = {zero, zero, zero, zero};
  unsigned vals[16];

  gather_im2col(vals, src, n0, 0, 1, tid);
  store_btile_row(ldsB[0], vals, tid);

  for (int chunk = 0; chunk < NCHK; ++chunk) {
    __syncthreads();
    bool more = (chunk + 1 < NCHK);
    if (more) gather_im2col(vals, src, n0, chunk + 1, 1, tid);   // overlap w/ WMMA
    v16bf a = load_frag_row(arowbase + (chunk << 5));
    if (more) __builtin_prefetch((const void*)(arowbase + ((chunk + 1) << 5)), 0, 3);
    const bf16_t* cur = ldsB[chunk & 1];
#pragma unroll
    for (int t = 0; t < 4; ++t) {
      v16bf bf = load_frag_row(cur + (nbase + (t << 4) + (lane & 15)) * LDSK +
                               ((lane >> 4) << 3));
      acc[t] = wmma_bf16(a, bf, acc[t]);
    }
    if (more) store_btile_row(ldsB[(chunk + 1) & 1], vals, tid);
  }
  const int hi = lane >> 4, nn = lane & 15;
#pragma unroll
  for (int t = 0; t < 4; ++t) {
    int n = n0 + nbase + (t << 4) + nn;
#pragma unroll
    for (int r = 0; r < 8; ++r) {
      int o = m0 + (hi << 3) + r;
      if (o < 27) om[((size_t)b * 27 + o) * HWSZ + n] = acc[t][r] + bias[o];
    }
  }
}

// ---------------- DCNv2: bilinear-gather cols -> WMMA, BN1+ReLU --------------
__global__ void __launch_bounds__(128) dcn_kernel(const float* __restrict__ x,
                                                  const float* __restrict__ om,
                                                  const bf16_t* __restrict__ wb,
                                                  const float* __restrict__ bias,
                                                  const float* __restrict__ s1,
                                                  const float* __restrict__ t1,
                                                  bf16_t* __restrict__ act1) {
  __shared__ float  sw[9 * NPOS * 4];     // mask*bilinear weight per corner
  __shared__ int    si[9 * NPOS * 4];     // clamped flat spatial index per corner
  __shared__ bf16_t ldsB[2][NPOS * LDSK];
  const int tid = threadIdx.x, lane = tid & 31, wave = tid >> 5;
  const int m0 = wave << 4;
  const int b  = blockIdx.y;
  const int n0 = blockIdx.x << 7;
  const float* xb_ = x + (size_t)b * CCH * HWSZ;
  const bf16_t* arowbase = wb + (size_t)(m0 + (lane & 15)) * KK + ((lane >> 4) << 3);

  // per-(tap, position) sampling setup from offset/mask conv output
  for (int t = tid; t < 9 * NPOS; t += 128) {
    int kpos = t >> 7, n = t & 127;
    int pos  = n0 + n;
    size_t base27 = (size_t)b * 27 * HWSZ + pos;
    float o1 = om[base27 + (size_t)kpos * HWSZ];
    float o2 = om[base27 + (size_t)(9 + kpos) * HWSZ];
    float ml = om[base27 + (size_t)(18 + kpos) * HWSZ];
    float mask = 1.f / (1.f + __expf(-ml));
    float py = (float)(pos >> 8) + (float)(kpos / 3 - 1) + o1;
    float px = (float)(pos & 255) + (float)(kpos % 3 - 1) + o2;
    float y0f = floorf(py), x0f = floorf(px);
    float wy1 = py - y0f, wx1 = px - x0f;
    int y0 = (int)y0f, x0 = (int)x0f, y1 = y0 + 1, x1 = x0 + 1;
    float vy0 = ((unsigned)y0 < 256u) ? 1.f : 0.f;
    float vy1 = ((unsigned)y1 < 256u) ? 1.f : 0.f;
    float vx0 = ((unsigned)x0 < 256u) ? 1.f : 0.f;
    float vx1 = ((unsigned)x1 < 256u) ? 1.f : 0.f;
    int cy0 = min(max(y0, 0), 255), cy1 = min(max(y1, 0), 255);
    int cx0 = min(max(x0, 0), 255), cx1 = min(max(x1, 0), 255);
    int e = t << 2;
    sw[e + 0] = mask * (1.f - wy1) * (1.f - wx1) * vy0 * vx0;
    sw[e + 1] = mask * (1.f - wy1) * wx1 * vy0 * vx1;
    sw[e + 2] = mask * wy1 * (1.f - wx1) * vy1 * vx0;
    sw[e + 3] = mask * wy1 * wx1 * vy1 * vx1;
    si[e + 0] = (cy0 << 8) + cx0;
    si[e + 1] = (cy0 << 8) + cx1;
    si[e + 2] = (cy1 << 8) + cx0;
    si[e + 3] = (cy1 << 8) + cx1;
  }
  __syncthreads();

  v8f zero = {0.f, 0.f, 0.f, 0.f, 0.f, 0.f, 0.f, 0.f};
  v8f acc[8] = {zero, zero, zero, zero, zero, zero, zero, zero};
  unsigned vals[16];

  // register-staged bilinear gather of one position's 32 channels for `chunk`
  auto gather_dcn = [&](unsigned out[16], int chunk) {
    int kpos = chunk >> 1, c0 = (chunk & 1) << 5;
    int e = ((kpos << 7) + tid) << 2;
    float w0 = sw[e + 0], w1 = sw[e + 1], w2 = sw[e + 2], w3 = sw[e + 3];
    int   i0 = si[e + 0], i1 = si[e + 1], i2 = si[e + 2], i3 = si[e + 3];
#pragma unroll
    for (int kk = 0; kk < 16; ++kk) {
      const float* xc0 = xb_ + (size_t)(c0 + 2 * kk) * HWSZ;
      const float* xc1 = xc0 + HWSZ;
      float v0 = w0 * xc0[i0] + w1 * xc0[i1] + w2 * xc0[i2] + w3 * xc0[i3];
      float v1 = w0 * xc1[i0] + w1 * xc1[i1] + w2 * xc1[i2] + w3 * xc1[i3];
      out[kk] = pack_bf2(v0, v1);
    }
  };

  gather_dcn(vals, 0);
  store_btile_row(ldsB[0], vals, tid);

  for (int chunk = 0; chunk < NCHK; ++chunk) {
    __syncthreads();
    bool more = (chunk + 1 < NCHK);
    if (more) gather_dcn(vals, chunk + 1);                       // overlap w/ WMMA
    v16bf a = load_frag_row(arowbase + (chunk << 5));
    if (more) __builtin_prefetch((const void*)(arowbase + ((chunk + 1) << 5)), 0, 3);
    const bf16_t* cur = ldsB[chunk & 1];
#pragma unroll
    for (int t = 0; t < 8; ++t) {
      v16bf bf = load_frag_row(cur + ((t << 4) + (lane & 15)) * LDSK +
                               ((lane >> 4) << 3));
      acc[t] = wmma_bf16(a, bf, acc[t]);
    }
    if (more) store_btile_row(ldsB[(chunk + 1) & 1], vals, tid);
  }
  const int hi = lane >> 4, nn = lane & 15;
#pragma unroll
  for (int t = 0; t < 8; ++t) {
    int n = n0 + (t << 4) + nn;
#pragma unroll
    for (int r = 0; r < 8; ++r) {
      int o = m0 + (hi << 3) + r;
      float v = (acc[t][r] + bias[o]) * s1[o] + t1[o];
      act1[((size_t)b * CCH + o) * HWSZ + n] = f2bf(fmaxf(v, 0.f));
    }
  }
}

// ---------------- dilated 3x3 conv 64->64: M=64, N-tile 128, BN+ReLU ---------
__global__ void __launch_bounds__(128) gemm_dil_kernel(const bf16_t* __restrict__ in,
                                                       const bf16_t* __restrict__ wb,
                                                       const float* __restrict__ bias,
                                                       const float* __restrict__ s,
                                                       const float* __restrict__ t,
                                                       bf16_t* __restrict__ outp,
                                                       int dil) {
  __shared__ bf16_t ldsB[2][NPOS * LDSK];
  const int tid = threadIdx.x, lane = tid & 31, wave = tid >> 5;
  const int m0 = wave << 4;
  const int b  = blockIdx.y;
  const int n0 = blockIdx.x << 7;
  const bf16_t* src = in + (size_t)b * CCH * HWSZ;
  const bf16_t* arowbase = wb + (size_t)(m0 + (lane & 15)) * KK + ((lane >> 4) << 3);

  v8f zero = {0.f, 0.f, 0.f, 0.f, 0.f, 0.f, 0.f, 0.f};
  v8f acc[8] = {zero, zero, zero, zero, zero, zero, zero, zero};
  unsigned vals[16];

  gather_im2col(vals, src, n0, 0, dil, tid);
  store_btile_row(ldsB[0], vals, tid);

  for (int chunk = 0; chunk < NCHK; ++chunk) {
    __syncthreads();
    bool more = (chunk + 1 < NCHK);
    if (more) gather_im2col(vals, src, n0, chunk + 1, dil, tid); // overlap w/ WMMA
    v16bf a = load_frag_row(arowbase + (chunk << 5));
    if (more) __builtin_prefetch((const void*)(arowbase + ((chunk + 1) << 5)), 0, 3);
    const bf16_t* cur = ldsB[chunk & 1];
#pragma unroll
    for (int tt = 0; tt < 8; ++tt) {
      v16bf bf = load_frag_row(cur + ((tt << 4) + (lane & 15)) * LDSK +
                               ((lane >> 4) << 3));
      acc[tt] = wmma_bf16(a, bf, acc[tt]);
    }
    if (more) store_btile_row(ldsB[(chunk + 1) & 1], vals, tid);
  }
  const int hi = lane >> 4, nn = lane & 15;
#pragma unroll
  for (int tt = 0; tt < 8; ++tt) {
    int n = n0 + (tt << 4) + nn;
#pragma unroll
    for (int r = 0; r < 8; ++r) {
      int o = m0 + (hi << 3) + r;
      float v = (acc[tt][r] + bias[o]) * s[o] + t[o];
      outp[((size_t)b * CCH + o) * HWSZ + n] = f2bf(fmaxf(v, 0.f));
    }
  }
}

// ---------------- final 64->1 conv + sigmoid + clip --------------------------
// Halo handled by clamped address + zero tap weight (no divergent branches).
__global__ void __launch_bounds__(256) final_conv_kernel(const bf16_t* __restrict__ act3,
                                                         const float* __restrict__ w3,
                                                         const float* __restrict__ b3,
                                                         float* __restrict__ out) {
  int i = blockIdx.x * 256 + threadIdx.x;
  int b = i >> 16, pos = i & 65535;
  int y = pos >> 8, x = pos & 255;
  float acc = b3[0];
#pragma unroll
  for (int kpos = 0; kpos < 9; ++kpos) {
    int yy = y + kpos / 3 - 1, xx = x + kpos % 3 - 1;
    float valid = ((unsigned)yy < 256u && (unsigned)xx < 256u) ? 1.f : 0.f;
    int yyc = min(max(yy, 0), 255), xxc = min(max(xx, 0), 255);
    int sp = (yyc << 8) + xxc;
    const bf16_t* a = act3 + (size_t)b * CCH * HWSZ + sp;
    for (int c = 0; c < CCH; ++c)
      acc += bf2f(a[(size_t)c * HWSZ]) * (w3[c * 9 + kpos] * valid);
  }
  float v = 1.f / (1.f + __expf(-acc));
  out[i] = fminf(fmaxf(v, 1e-4f), 1.f - 1e-4f);
}

// ---------------- host launcher ----------------
extern "C" void kernel_launch(void* const* d_in, const int* in_sizes, int n_in,
                              void* d_out, int out_size, void* d_ws, size_t ws_size,
                              hipStream_t stream) {
  const float* x     = (const float*)d_in[0];
  const float* w_om  = (const float*)d_in[1];
  const float* b_om  = (const float*)d_in[2];
  const float* w_dcn = (const float*)d_in[3];
  const float* b_dcn = (const float*)d_in[4];
  const float* bn1g = (const float*)d_in[5],  *bn1b = (const float*)d_in[6];
  const float* bn1m = (const float*)d_in[7],  *bn1v = (const float*)d_in[8];
  const float* bn2g = (const float*)d_in[9],  *bn2b = (const float*)d_in[10];
  const float* bn2m = (const float*)d_in[11], *bn2v = (const float*)d_in[12];
  const float* bn3g = (const float*)d_in[13], *bn3b = (const float*)d_in[14];
  const float* bn3m = (const float*)d_in[15], *bn3v = (const float*)d_in[16];
  const float* w_h = (const float*)d_in[17], *b_h = (const float*)d_in[18];
  const float* w_w = (const float*)d_in[19], *b_w = (const float*)d_in[20];
  const float* w3  = (const float*)d_in[21], *b3  = (const float*)d_in[22];

  char* ws = (char*)d_ws;
  bf16_t* xb    = (bf16_t*)(ws + 0);           // 16.78 MB
  bf16_t* act1  = (bf16_t*)(ws + 16777216);    // 16.78 MB
  bf16_t* act2  = (bf16_t*)(ws + 33554432);    // 16.78 MB
  bf16_t* act3  = (bf16_t*)(ws + 50331648);    // 16.78 MB
  float*  om    = (float*) (ws + 67108864);    // 14.16 MB
  bf16_t* wbom  = (bf16_t*)(ws + 81264640);    // 36 KB
  bf16_t* wbdcn = (bf16_t*)(ws + 81301504);    // 72 KB
  bf16_t* wbh   = (bf16_t*)(ws + 81375232);    // 72 KB
  bf16_t* wbw   = (bf16_t*)(ws + 81448960);    // 72 KB
  float*  bn    = (float*) (ws + 81522688);    // 1.5 KB

  cvt_kernel<<<8192, 256, 0, stream>>>(x, xb);
  pack_kernel<<<(32 * KK + 3 * 64 * KK + 192 + 255) / 256, 256, 0, stream>>>(
      w_om, w_dcn, w_h, w_w,
      bn1g, bn1b, bn1m, bn1v, bn2g, bn2b, bn2m, bn2v, bn3g, bn3b, bn3m, bn3v,
      wbom, wbdcn, wbh, wbw, bn);

  dim3 grid(HWSZ / NPOS, 2);   // (512, 2)
  gemm_om_kernel<<<grid, 128, 0, stream>>>(xb, wbom, b_om, om);
  dcn_kernel<<<grid, 128, 0, stream>>>(x, om, wbdcn, b_dcn, bn + 0, bn + 64, act1);
  gemm_dil_kernel<<<grid, 128, 0, stream>>>(act1, wbh, b_h, bn + 128, bn + 192, act2, 2);
  gemm_dil_kernel<<<grid, 128, 0, stream>>>(act2, wbw, b_w, bn + 256, bn + 320, act3, 2);
  final_conv_kernel<<<(2 * HWSZ) / 256, 256, 0, stream>>>(act3, w3, b3, (float*)d_out);
}
